// PointNetPlusPlusCls_SSG_5549097746742
// MI455X (gfx1250) — compile-verified, matrix operand order checked by reading, not run on hardware
//
#include <hip/hip_runtime.h>
#include <hip/hip_bf16.h>
#include <hip/hip_fp16.h>

typedef __attribute__((ext_vector_type(16))) _Float16 v16h;
typedef __attribute__((ext_vector_type(8)))  _Float16 v8h;
typedef __attribute__((ext_vector_type(8)))  float    v8f;

#define BN_EPS 1e-5f

// ---------------------------------------------------------------- kernels

__global__ void k_split_input(const float* __restrict__ pc, float* __restrict__ xyz,
                              float* __restrict__ nrm, int Bv, int Nv) {
  int i = blockIdx.x * blockDim.x + threadIdx.x;
  if (i >= Bv * Nv) return;
  int b = i / Nv, n = i % Nv;
  const float* src = pc + (size_t)b * 6 * Nv;
  float* x = xyz + (size_t)(b * Nv + n) * 3;
  float* m = nrm + (size_t)(b * Nv + n) * 3;
  x[0] = src[0 * Nv + n]; x[1] = src[1 * Nv + n]; x[2] = src[2 * Nv + n];
  m[0] = src[3 * Nv + n]; m[1] = src[4 * Nv + n]; m[2] = src[5 * Nv + n];
}

// Farthest point sampling: one block per batch, dist array in LDS.
__global__ void k_fps(const float* __restrict__ xyz, int Nv, int npoint,
                      int* __restrict__ outidx) {
  __shared__ float dist[4096];
  __shared__ float rv[256];
  __shared__ int   ri[256];
  __shared__ int   s_far;
  int b = blockIdx.x;
  int t = threadIdx.x;
  const float* base = xyz + (size_t)b * Nv * 3;
  for (int i = t; i < Nv; i += 256) dist[i] = 1e10f;
  if (t == 0) s_far = 0;
  __syncthreads();
  for (int it = 0; it < npoint; ++it) {
    int far = s_far;
    if (t == 0) outidx[b * npoint + it] = far;
    float cx = base[far * 3 + 0], cy = base[far * 3 + 1], cz = base[far * 3 + 2];
    float bv = -1.0f; int bi = Nv;
    for (int i = t; i < Nv; i += 256) {
      float dx = base[i * 3 + 0] - cx;
      float dy = base[i * 3 + 1] - cy;
      float dz = base[i * 3 + 2] - cz;
      float d = dx * dx + dy * dy + dz * dz;
      float dm = dist[i];
      if (d < dm) { dm = d; dist[i] = dm; }
      if (dm > bv) { bv = dm; bi = i; }   // keeps lowest index on ties within stride
    }
    rv[t] = bv; ri[t] = bi;
    __syncthreads();
    for (int s = 128; s > 0; s >>= 1) {
      if (t < s) {
        float v2 = rv[t + s]; int i2 = ri[t + s];
        if (v2 > rv[t] || (v2 == rv[t] && i2 < ri[t])) { rv[t] = v2; ri[t] = i2; }
      }
      __syncthreads();
    }
    if (t == 0) s_far = ri[0];
    __syncthreads();
  }
}

__global__ void k_gather3(const float* __restrict__ xyz, const int* __restrict__ idx,
                          float* __restrict__ out, int Bv, int Sv, int Nv) {
  int i = blockIdx.x * blockDim.x + threadIdx.x;
  if (i >= Bv * Sv) return;
  int b = i / Sv, s = i % Sv;
  int g = idx[b * Sv + s];
  const float* p = xyz + (size_t)(b * Nv + g) * 3;
  float* o = out + (size_t)i * 3;
  o[0] = p[0]; o[1] = p[1]; o[2] = p[2];
}

// First `ns` indices (ascending) within radius; pad with first found.
__global__ void k_ballquery(const float* __restrict__ xyz, const float* __restrict__ newxyz,
                            int* __restrict__ gidx, int Bv, int Sv, int Nv, int ns, float r2) {
  int t = blockIdx.x * blockDim.x + threadIdx.x;
  if (t >= Bv * Sv) return;
  int b = t / Sv;
  const float* base = xyz + (size_t)b * Nv * 3;
  float cx = newxyz[(size_t)t * 3 + 0];
  float cy = newxyz[(size_t)t * 3 + 1];
  float cz = newxyz[(size_t)t * 3 + 2];
  int* out = gidx + (size_t)t * ns;
  int cnt = 0;
  for (int j = 0; j < Nv && cnt < ns; ++j) {
    float dx = base[j * 3 + 0] - cx;
    float dy = base[j * 3 + 1] - cy;
    float dz = base[j * 3 + 2] - cz;
    if (dx * dx + dy * dy + dz * dz <= r2) out[cnt++] = j;
  }
  int f = (cnt > 0) ? out[0] : 0;
  for (; cnt < ns; ++cnt) out[cnt] = f;
}

// Build grouped features row-major f16 (padded K): [xyz_diff(3), pts(Cpts), 0-pad]
__global__ void k_build_grouped(const float* __restrict__ xyz, const float* __restrict__ newxyz,
                                const float* __restrict__ pts, const int* __restrict__ gidx,
                                _Float16* __restrict__ out,
                                int Bv, int Sv, int ns, int Nv, int Cpts, int Kp) {
  long long i = (long long)blockIdx.x * blockDim.x + threadIdx.x;
  long long total = (long long)Bv * Sv * ns * Kp;
  if (i >= total) return;
  int c = (int)(i % Kp);
  long long row = i / Kp;
  int k = (int)(row % ns);
  long long bs = row / ns;
  int s = (int)(bs % Sv);
  int b = (int)(bs / Sv);
  int g = gidx[(size_t)(b * Sv + s) * ns + k];
  float v = 0.f;
  if (c < 3)            v = xyz[((size_t)b * Nv + g) * 3 + c] - newxyz[((size_t)b * Sv + s) * 3 + c];
  else if (c < 3 + Cpts) v = pts[((size_t)b * Nv + g) * Cpts + (c - 3)];
  out[(size_t)row * Kp + c] = (_Float16)v;
}

// group_all: concat(xyz, pts) raw (no centering)
__global__ void k_build_groupall(const float* __restrict__ xyz, const float* __restrict__ pts,
                                 _Float16* __restrict__ out, int Bv, int Nv, int Cpts, int Kp) {
  long long i = (long long)blockIdx.x * blockDim.x + threadIdx.x;
  long long total = (long long)Bv * Nv * Kp;
  if (i >= total) return;
  int c = (int)(i % Kp);
  long long row = i / Kp;
  float v = 0.f;
  if (c < 3)             v = xyz[(size_t)row * 3 + c];
  else if (c < 3 + Cpts) v = pts[(size_t)row * Cpts + (c - 3)];
  out[(size_t)row * Kp + c] = (_Float16)v;
}

// W (Cout x Cin) f32 -> (Coutp x Kp) f16, zero padded
__global__ void k_wconv(const float* __restrict__ W, _Float16* __restrict__ out,
                        int Cout, int Cin, int Coutp, int Kp) {
  int i = blockIdx.x * blockDim.x + threadIdx.x;
  if (i >= Coutp * Kp) return;
  int co = i / Kp, k = i % Kp;
  out[i] = (co < Cout && k < Cin) ? (_Float16)W[(size_t)co * Cin + k] : (_Float16)0.f;
}

// Y(MxNp,f32) = A(MxKp,f16) * Wt(NpxKp,f16)^T via WMMA. M,Np multiples of 16, Kp of 32.
__global__ void k_gemm_wmma(const _Float16* __restrict__ A, const _Float16* __restrict__ Wt,
                            float* __restrict__ Y, int M, int Np, int Kp) {
  int wid  = (int)((blockIdx.x * blockDim.x + threadIdx.x) >> 5);
  int lane = threadIdx.x & 31;
  int ntiles = Np >> 4;
  int mtiles = M >> 4;
  if (wid >= mtiles * ntiles) return;          // wave-uniform exit, EXEC stays full
  int mt = wid / ntiles;
  int nt = wid % ntiles;
  int half = lane >> 4;                        // 0|1
  int l15  = lane & 15;
  int koff = half * 8;
  const _Float16* arow = A  + (size_t)(mt * 16 + l15) * Kp + koff;
  const _Float16* brow = Wt + (size_t)(nt * 16 + l15) * Kp + koff;
  v8f acc = {0.f, 0.f, 0.f, 0.f, 0.f, 0.f, 0.f, 0.f};
  for (int k = 0; k < Kp; k += 32) {
    v8h alo = *(const v8h*)(arow + k);
    v8h ahi = *(const v8h*)(arow + k + 16);
    v8h blo = *(const v8h*)(brow + k);
    v8h bhi = *(const v8h*)(brow + k + 16);
    v16h a  = __builtin_shufflevector(alo, ahi, 0,1,2,3,4,5,6,7,8,9,10,11,12,13,14,15);
    v16h bb = __builtin_shufflevector(blo, bhi, 0,1,2,3,4,5,6,7,8,9,10,11,12,13,14,15);
    acc = __builtin_amdgcn_wmma_f32_16x16x32_f16(false, a, false, bb, (short)0, acc, false, false);
  }
  int r0 = mt * 16 + half * 8;
  float* yp = Y + (size_t)r0 * Np + nt * 16 + l15;
#pragma unroll
  for (int r = 0; r < 8; ++r) yp[(size_t)r * Np] = acc[r];
}

// per-column BN stats -> stats[c]=g*rsqrt(var+eps), stats[Nc+c]=beta-mean*scale
__global__ void k_colstats(const float* __restrict__ Y, const float* __restrict__ g,
                           const float* __restrict__ beta, float* __restrict__ stats,
                           int M, int Nc, int ld) {
  __shared__ float s1[256], s2[256];
  int c = blockIdx.x;
  int t = threadIdx.x;
  float sum = 0.f, sq = 0.f;
  for (int r = t; r < M; r += 256) {
    float v = Y[(size_t)r * ld + c];
    sum += v; sq += v * v;
  }
  s1[t] = sum; s2[t] = sq;
  __syncthreads();
  for (int s = 128; s > 0; s >>= 1) {
    if (t < s) { s1[t] += s1[t + s]; s2[t] += s2[t + s]; }
    __syncthreads();
  }
  if (t == 0) {
    float m = s1[0] / (float)M;
    float v = s2[0] / (float)M - m * m;
    if (v < 0.f) v = 0.f;
    float sc = g[c] * rsqrtf(v + BN_EPS);
    stats[c] = sc;
    stats[Nc + c] = beta[c] - m * sc;
  }
}

__global__ void k_bnrelu_f16(const float* __restrict__ Y, const float* __restrict__ stats,
                             _Float16* __restrict__ out, int M, int Nc, int ldin, int Kpn) {
  long long i = (long long)blockIdx.x * blockDim.x + threadIdx.x;
  long long total = (long long)M * Kpn;
  if (i >= total) return;
  int c = (int)(i % Kpn);
  long long r = i / Kpn;
  float v = 0.f;
  if (c < Nc) {
    v = Y[(size_t)r * ldin + c] * stats[c] + stats[Nc + c];
    if (v < 0.f) v = 0.f;
  }
  out[i] = (_Float16)v;
}

__global__ void k_bnrelu_f32_inplace(float* __restrict__ Y, const float* __restrict__ stats,
                                     int M, int Nc, int ld) {
  long long i = (long long)blockIdx.x * blockDim.x + threadIdx.x;
  long long total = (long long)M * Nc;
  if (i >= total) return;
  int c = (int)(i % Nc);
  long long r = i / Nc;
  float v = Y[(size_t)r * ld + c] * stats[c] + stats[Nc + c];
  if (v < 0.f) v = 0.f;
  Y[(size_t)r * ld + c] = v;
}

__global__ void k_maxpool(const float* __restrict__ Y, float* __restrict__ out,
                          int rows_out, int ns, int C, int ld) {
  long long i = (long long)blockIdx.x * blockDim.x + threadIdx.x;
  long long total = (long long)rows_out * C;
  if (i >= total) return;
  int c = (int)(i % C);
  long long ro = i / C;
  const float* p = Y + (size_t)ro * ns * ld + c;
  float m = p[0];
  for (int k = 1; k < ns; ++k) {
    float v = p[(size_t)k * ld];
    if (v > m) m = v;
  }
  out[i] = m;
}

__global__ void k_f2h(const float* __restrict__ in, _Float16* __restrict__ out, long long n) {
  long long i = (long long)blockIdx.x * blockDim.x + threadIdx.x;
  if (i < n) out[i] = (_Float16)in[i];
}

__global__ void k_copy(const float* __restrict__ in, float* __restrict__ out, long long n) {
  long long i = (long long)blockIdx.x * blockDim.x + threadIdx.x;
  if (i < n) out[i] = in[i];
}

// cls3: add bias, log_softmax over 40 classes, write logits
__global__ void k_logits(const float* __restrict__ Y, const float* __restrict__ bias,
                         float* __restrict__ out) {
  __shared__ float red[64];
  __shared__ float vals[64];
  int b = blockIdx.x;
  int t = threadIdx.x;
  float v = -1e30f;
  if (t < 40) { v = Y[b * 48 + t] + bias[t]; vals[t] = v; }
  red[t] = v;
  __syncthreads();
  for (int s = 32; s > 0; s >>= 1) { if (t < s) red[t] = fmaxf(red[t], red[t + s]); __syncthreads(); }
  float mx = red[0];
  __syncthreads();
  float e = (t < 40) ? __expf(vals[t] - mx) : 0.f;
  red[t] = e;
  __syncthreads();
  for (int s = 32; s > 0; s >>= 1) { if (t < s) red[t] += red[t + s]; __syncthreads(); }
  float lse = logf(red[0]) + mx;
  if (t < 40) out[b * 40 + t] = vals[t] - lse;
}

// ---------------------------------------------------------------- host side

static inline int cdiv_ll(long long a, long long b) { return (int)((a + b - 1) / b); }

struct LayerP { const float* W; const float* b; const float* g; const float* beta; };

static void run_bn_layer(const _Float16* A, const _Float16* Wt, float* Y, float* stats,
                         const float* g, const float* beta,
                         int M, int Cout, int Coutp, int Kp,
                         _Float16* outH, int Kpn, hipStream_t stream) {
  long long tiles = (long long)(M / 16) * (Coutp / 16);
  k_gemm_wmma<<<cdiv_ll(tiles, 8), 256, 0, stream>>>(A, Wt, Y, M, Coutp, Kp);
  k_colstats<<<Cout, 256, 0, stream>>>(Y, g, beta, stats, M, Cout, Coutp);
  if (outH) {
    long long n = (long long)M * Kpn;
    k_bnrelu_f16<<<cdiv_ll(n, 256), 256, 0, stream>>>(Y, stats, outH, M, Cout, Coutp, Kpn);
  } else {
    long long n = (long long)M * Cout;
    k_bnrelu_f32_inplace<<<cdiv_ll(n, 256), 256, 0, stream>>>(Y, stats, M, Cout, Coutp);
  }
}

extern "C" void kernel_launch(void* const* d_in, const int* in_sizes, int n_in,
                              void* d_out, int out_size, void* d_ws, size_t ws_size,
                              hipStream_t stream) {
  const int Bv = 16, Nv = 4096;

  // ---- map parameters (handle both insertion-order and jax-sorted flatten) ----
  const float* pc = nullptr;
  LayerP L[11];            // 0..8 = sa1..sa3 layers, 9 = cls1, 10 = cls2
  const float* cls3W = nullptr;
  const float* cls3b = nullptr;
  if (in_sizes[0] == Bv * 6 * Nv) {
    // insertion order: point_cloud, sa1(3x4), sa2(3x4), sa3(3x4), cls1(4), cls2(4), cls3(2)
    pc = (const float*)d_in[0];
    int p = 1;
    for (int l = 0; l < 11; ++l) {
      L[l].W = (const float*)d_in[p]; L[l].b = (const float*)d_in[p + 1];
      L[l].g = (const float*)d_in[p + 2]; L[l].beta = (const float*)d_in[p + 3];
      p += 4;
    }
    cls3W = (const float*)d_in[p]; cls3b = (const float*)d_in[p + 1];
  } else {
    // sorted keys: cls1(4), cls2(4), cls3(2), sa1(12), sa2(12), sa3(12), point_cloud
    L[9]  = { (const float*)d_in[0], (const float*)d_in[1], (const float*)d_in[2], (const float*)d_in[3] };
    L[10] = { (const float*)d_in[4], (const float*)d_in[5], (const float*)d_in[6], (const float*)d_in[7] };
    cls3W = (const float*)d_in[8]; cls3b = (const float*)d_in[9];
    int p = 10;
    for (int l = 0; l < 9; ++l) {
      L[l].W = (const float*)d_in[p]; L[l].b = (const float*)d_in[p + 1];
      L[l].g = (const float*)d_in[p + 2]; L[l].beta = (const float*)d_in[p + 3];
      p += 4;
    }
    pc = (const float*)d_in[p];
  }

  // ---- layer geometry ----
  static const int cin[12]   = {6, 64, 64, 131, 128, 128, 259, 256, 512, 1024, 512, 256};
  static const int cout[12]  = {64, 64, 128, 128, 128, 256, 256, 512, 1024, 512, 256, 40};
  static const int kp[12]    = {32, 64, 64, 160, 128, 128, 288, 256, 512, 1024, 512, 256};
  static const int coutp[12] = {64, 64, 128, 128, 128, 256, 256, 512, 1024, 512, 256, 48};
  static const long long whoff[12] = {0, 2048, 6144, 14336, 34816, 51200, 83968,
                                      157696, 288768, 813056, 1337344, 1468416};

  // ---- workspace layout (bytes) ----
  char* ws = (char*)d_ws;
  const size_t WS_XYZ   = 0;                          // 16*4096*3*4  = 786432
  const size_t WS_NRM   = WS_XYZ  + 786432;
  const size_t WS_XYZ1  = WS_NRM  + 786432;           // 16*512*3*4   = 98304
  const size_t WS_XYZ2  = WS_XYZ1 + 98304;            // 16*128*3*4   = 24576
  const size_t WS_FPS   = WS_XYZ2 + 24576;            // 16*512*4     = 32768
  const size_t WS_GIDX  = WS_FPS  + 32768;            // 16*512*32*4  = 1048576
  const size_t WS_PTS1  = WS_GIDX + 1048576;          // 16*512*128*4 = 4194304
  const size_t WS_PTS2  = WS_PTS1 + 4194304;          // 16*128*256*4 = 2097152, 524288 reserved
  const size_t WS_FEAT  = WS_PTS2 + 524288;           // 16*1024*4    = 65536
  const size_t WS_STATS = WS_FEAT + 65536;            // 2*1024*4     = 8192
  const size_t WS_WH    = WS_STATS + 8192;            // 1480704 halves = 2961408
  const size_t WS_A     = WS_WH + 2961408;            // 262144*64*2  = 33554432
  const size_t WS_B     = WS_A  + 33554432;           // 33554432
  const size_t WS_Y     = WS_B  + 33554432;           // 262144*128*4 = 134217728

  float*     xyz    = (float*)(ws + WS_XYZ);
  float*     nrm    = (float*)(ws + WS_NRM);
  float*     xyz1   = (float*)(ws + WS_XYZ1);
  float*     xyz2   = (float*)(ws + WS_XYZ2);
  int*       fpsidx = (int*)(ws + WS_FPS);
  int*       gidx   = (int*)(ws + WS_GIDX);
  float*     pts1   = (float*)(ws + WS_PTS1);
  float*     pts2   = (float*)(ws + WS_PTS2);
  float*     feat   = (float*)(ws + WS_FEAT);
  float*     stats  = (float*)(ws + WS_STATS);
  _Float16*  wh     = (_Float16*)(ws + WS_WH);
  _Float16*  Abuf   = (_Float16*)(ws + WS_A);
  _Float16*  Bbuf   = (_Float16*)(ws + WS_B);
  float*     Ybuf   = (float*)(ws + WS_Y);

  // ---- convert weights to padded f16 ----
  for (int l = 0; l < 11; ++l) {
    int n = coutp[l] * kp[l];
    k_wconv<<<cdiv_ll(n, 256), 256, 0, stream>>>(L[l].W, wh + whoff[l], cout[l], cin[l], coutp[l], kp[l]);
  }
  k_wconv<<<cdiv_ll(48 * 256, 256), 256, 0, stream>>>(cls3W, wh + whoff[11], 40, 256, 48, 256);

  // ---- split input ----
  k_split_input<<<cdiv_ll((long long)Bv * Nv, 256), 256, 0, stream>>>(pc, xyz, nrm, Bv, Nv);

  // ======== SA1: N=4096 -> S=512, ns=32, r=0.2 ========
  const int S1 = 512, NS1 = 32;
  const long long M1 = (long long)Bv * S1 * NS1;  // 262144
  k_fps<<<Bv, 256, 0, stream>>>(xyz, Nv, S1, fpsidx);
  k_gather3<<<cdiv_ll((long long)Bv * S1, 256), 256, 0, stream>>>(xyz, fpsidx, xyz1, Bv, S1, Nv);
  k_ballquery<<<cdiv_ll((long long)Bv * S1, 256), 256, 0, stream>>>(xyz, xyz1, gidx, Bv, S1, Nv, NS1, 0.04f);
  k_build_grouped<<<cdiv_ll(M1 * 32, 256), 256, 0, stream>>>(xyz, xyz1, nrm, gidx, Abuf,
                                                             Bv, S1, NS1, Nv, 3, 32);
  run_bn_layer(Abuf, wh + whoff[0], Ybuf, stats, L[0].g, L[0].beta, (int)M1, 64, 64, 32, Bbuf, 64, stream);
  run_bn_layer(Bbuf, wh + whoff[1], Ybuf, stats, L[1].g, L[1].beta, (int)M1, 64, 64, 64, Abuf, 64, stream);
  run_bn_layer(Abuf, wh + whoff[2], Ybuf, stats, L[2].g, L[2].beta, (int)M1, 128, 128, 64, nullptr, 0, stream);
  k_maxpool<<<cdiv_ll((long long)Bv * S1 * 128, 256), 256, 0, stream>>>(Ybuf, pts1, Bv * S1, NS1, 128, 128);

  // ======== SA2: N=512 -> S=128, ns=64, r=0.4 ========
  const int S2 = 128, NS2 = 64;
  const long long M2 = (long long)Bv * S2 * NS2;  // 131072
  k_fps<<<Bv, 256, 0, stream>>>(xyz1, S1, S2, fpsidx);
  k_gather3<<<cdiv_ll((long long)Bv * S2, 256), 256, 0, stream>>>(xyz1, fpsidx, xyz2, Bv, S2, S1);
  k_ballquery<<<cdiv_ll((long long)Bv * S2, 256), 256, 0, stream>>>(xyz1, xyz2, gidx, Bv, S2, S1, NS2, 0.16f);
  k_build_grouped<<<cdiv_ll(M2 * 160, 256), 256, 0, stream>>>(xyz1, xyz2, pts1, gidx, Abuf,
                                                              Bv, S2, NS2, S1, 128, 160);
  run_bn_layer(Abuf, wh + whoff[3], Ybuf, stats, L[3].g, L[3].beta, (int)M2, 128, 128, 160, Bbuf, 128, stream);
  run_bn_layer(Bbuf, wh + whoff[4], Ybuf, stats, L[4].g, L[4].beta, (int)M2, 128, 128, 128, Abuf, 128, stream);
  run_bn_layer(Abuf, wh + whoff[5], Ybuf, stats, L[5].g, L[5].beta, (int)M2, 256, 256, 128, nullptr, 0, stream);
  k_maxpool<<<cdiv_ll((long long)Bv * S2 * 256, 256), 256, 0, stream>>>(Ybuf, pts2, Bv * S2, NS2, 256, 256);

  // ======== SA3: group all (128 points, 259 ch) ========
  const long long M3 = (long long)Bv * S2;        // 2048
  k_build_groupall<<<cdiv_ll(M3 * 288, 256), 256, 0, stream>>>(xyz2, pts2, Abuf, Bv, S2, 256, 288);
  run_bn_layer(Abuf, wh + whoff[6], Ybuf, stats, L[6].g, L[6].beta, (int)M3, 256, 256, 288, Bbuf, 256, stream);
  run_bn_layer(Bbuf, wh + whoff[7], Ybuf, stats, L[7].g, L[7].beta, (int)M3, 512, 512, 256, Abuf, 512, stream);
  run_bn_layer(Abuf, wh + whoff[8], Ybuf, stats, L[8].g, L[8].beta, (int)M3, 1024, 1024, 512, nullptr, 0, stream);
  k_maxpool<<<cdiv_ll((long long)Bv * 1024, 256), 256, 0, stream>>>(Ybuf, feat, Bv, S2, 1024, 1024);

  // ======== classifier head (M=16) ========
  k_f2h<<<cdiv_ll(16 * 1024, 256), 256, 0, stream>>>(feat, Abuf, 16 * 1024);
  run_bn_layer(Abuf, wh + whoff[9],  Ybuf, stats, L[9].g,  L[9].beta,  16, 512, 512, 1024, Bbuf, 512, stream);
  run_bn_layer(Bbuf, wh + whoff[10], Ybuf, stats, L[10].g, L[10].beta, 16, 256, 256, 512, Abuf, 256, stream);
  {
    long long tiles = (16 / 16) * (48 / 16);  // 3 tiles
    k_gemm_wmma<<<cdiv_ll(tiles, 8), 256, 0, stream>>>(Abuf, wh + whoff[11], Ybuf, 16, 48, 256);
  }
  k_logits<<<16, 64, 0, stream>>>(Ybuf, cls3b, (float*)d_out);
  k_copy<<<cdiv_ll(16 * 1024, 256), 256, 0, stream>>>(feat, (float*)d_out + 640, 16 * 1024);

  (void)n_in; (void)out_size; (void)ws_size;
}
